// Module_47553877901520
// MI455X (gfx1250) — matrix operand order repaired and checked by reading, not compiled, not run on hardware
//
#include <hip/hip_runtime.h>
#include <hip/hip_bf16.h>
#include <stdint.h>

// ---- model dims ----
#define Bz 8
#define Sz 256
#define Tz 64
#define Ez 128
#define Hz 256
#define Vz 50000
#define MPz 16
#define Gz 768          // 3*H
#define POS_SZ 2048     // MP*E
#define POS_H 1024      // POS_SZ/2
#define DEC_IN 640      // 3*E + H
#define EOUT 512        // 2*H
#define Mz (Tz * Bz)    // 512 decoder rows

typedef __bf16 bf16_t;
typedef __attribute__((ext_vector_type(16))) __bf16 v16bf;
typedef __attribute__((ext_vector_type(8)))  float  v8f;
// global-address-space weight pointer: survives asm laundering without the
// backend losing addrspace info (keeps global_load_b128, not flat_load).
typedef __attribute__((address_space(1))) const bf16_t* gbf_ptr;

__device__ __forceinline__ float sigf(float x) { return 1.0f / (1.0f + __expf(-x)); }

// WMMA fragment loaders (contiguous k = 16h + s order, same for A and B).
// A 16x32: lanes 0-15 -> M=lane, elems 0..15 = K 0..15;
//          lanes 16-31 -> M=lane-16, K offsets +16.  Caller pre-adds (hi?16:0)+k0.
__device__ __forceinline__ v16bf load_frag_a(const bf16_t* p) {
  v16bf a;
#pragma unroll
  for (int e = 0; e < 16; ++e) a[e] = p[e];
  return a;
}
// B 32x16: lane -> N=lane&15; 16 consecutive K per lane (lanes>=16: +16). Caller pre-adds.
__device__ __forceinline__ v16bf load_frag_b(const bf16_t* p) {
  v16bf b;
#pragma unroll
  for (int e = 0; e < 16; ++e) b[e] = p[e];
  return b;
}
__device__ __forceinline__ v16bf load_frag_b_g(gbf_ptr p) {
  v16bf b;
#pragma unroll
  for (int e = 0; e < 16; ++e) b[e] = p[e];
  return b;
}

// ---------------- gathers ----------------
__global__ void gather_k(const int* __restrict__ idx, const float* __restrict__ tab,
                         float* __restrict__ out, int n, int ed) {
  int i = blockIdx.x * blockDim.x + threadIdx.x;
  if (i >= n * ed) return;
  int r = i / ed, e = i % ed;
  out[i] = tab[(size_t)idx[r] * ed + e];
}

// ---------------- generic fc + tanh ----------------
__global__ void fc_tanh_k(const float* __restrict__ x, const float* __restrict__ W,
                          const float* __restrict__ bias, float* __restrict__ y,
                          int in_dim, int out_dim, int total) {
  int i = blockIdx.x * blockDim.x + threadIdx.x;
  if (i >= total) return;
  int r = i / out_dim, o = i % out_dim;
  const float* xr = x + (size_t)r * in_dim;
  const float* wr = W + (size_t)o * in_dim;
  float s = bias[o];
  for (int k = 0; k < in_dim; ++k) s += xr[k] * wr[k];
  y[i] = tanhf(s);
}

// ---------------- encoder input gates ----------------
__global__ void enc_gi_k(const float* __restrict__ x_emb,
                         const float* __restrict__ Wf, const float* __restrict__ bf_,
                         const float* __restrict__ Wb, const float* __restrict__ bb_,
                         float* __restrict__ gf, float* __restrict__ gb) {
  const int TOT = 2 * Bz * Sz * Gz;
  int idx = blockIdx.x * blockDim.x + threadIdx.x;
  if (idx >= TOT) return;
  int dir = idx >= Bz * Sz * Gz;
  int r = idx - dir * Bz * Sz * Gz;
  int bs = r / Gz, g = r % Gz;
  const float* x = x_emb + (size_t)bs * Ez;
  const float* W = (dir ? Wb : Wf) + (size_t)g * Ez;
  float s = (dir ? bb_ : bf_)[g];
  for (int k = 0; k < Ez; ++k) s += x[k] * W[k];
  (dir ? gb : gf)[r] = s;
}

// ---------------- f32 -> bf16 convert ----------------
__global__ void tobf16_k(const float* __restrict__ x, bf16_t* __restrict__ y, long long n) {
  long long i = blockIdx.x * (long long)blockDim.x + threadIdx.x;
  long long stride = (long long)gridDim.x * blockDim.x;
  for (; i < n; i += stride) y[i] = (bf16_t)x[i];
}

// =======================================================================
// GRU recurrence step core, shared by encoder/decoder scans.
// 512 threads = 16 waves; wave w owns gate-row tiles 3w..3w+2 (48 tiles).
// kb-outer / tile-inner with 3 live accumulators: one LDS A-fragment load
// per k-block feeds 3 WMMAs; 3 independent chains hide D->A/B hazards.
// Weight pointer is laundered per step (asm, addrspace(1) preserved) so
// LICM cannot hoist the 192-VGPR fragment set and spill it to scratch.
// =======================================================================
__device__ __forceinline__ void gru_gh_wmma(gbf_ptr Wp,
                                            const bf16_t* hbf, float* gh,
                                            int wave, int n, bool hi) {
  const int nt0 = wave * 3;
  const int row0 = nt0 * 16 + n;
  v8f c0 = {}, c1 = {}, c2 = {};
  const bf16_t* arow = hbf + (size_t)n * Hz + (hi ? 16 : 0);
  gbf_ptr b0p = Wp + (size_t)row0 * Hz + (hi ? 16 : 0);
  gbf_ptr b1p = b0p + (size_t)16 * Hz;
  gbf_ptr b2p = b1p + (size_t)16 * Hz;
#pragma unroll
  for (int kb = 0; kb < Hz / 32; ++kb) {
    const int k0 = kb * 32;
    v16bf a  = load_frag_a(arow + k0);
    v16bf b0 = load_frag_b_g(b0p + k0);
    v16bf b1 = load_frag_b_g(b1p + k0);
    v16bf b2 = load_frag_b_g(b2p + k0);
    c0 = __builtin_amdgcn_wmma_f32_16x16x32_bf16(false, a, false, b0, (short)0, c0, false, false);
    c1 = __builtin_amdgcn_wmma_f32_16x16x32_bf16(false, a, false, b1, (short)0, c1, false, false);
    c2 = __builtin_amdgcn_wmma_f32_16x16x32_bf16(false, a, false, b2, (short)0, c2, false, false);
  }
  if (!hi) {  // lanes 0-15: c[r] = batch r, gate row = tile*16 + n
#pragma unroll
    for (int r = 0; r < 8; ++r) gh[(row0)      * Bz + r] = c0[r];
#pragma unroll
    for (int r = 0; r < 8; ++r) gh[(row0 + 16) * Bz + r] = c1[r];
#pragma unroll
    for (int r = 0; r < 8; ++r) gh[(row0 + 32) * Bz + r] = c2[r];
  }
}

__global__ void __launch_bounds__(512) enc_scan_wmma_k(
    const float* __restrict__ gi_f, const float* __restrict__ gi_b,
    const bf16_t* __restrict__ Whf, const bf16_t* __restrict__ Whb,
    const float* __restrict__ bhh_f, const float* __restrict__ bhh_b,
    float* __restrict__ enc_out, float* __restrict__ hfin_f, float* __restrict__ hfin_b) {
  const int dir = blockIdx.x;
  const float*  gi  = dir ? gi_b : gi_f;
  const float*  bhh = dir ? bhh_b : bhh_f;
  float* hfin = dir ? hfin_b : hfin_f;
  gbf_ptr wp = (gbf_ptr)(dir ? Whb : Whf);

  __shared__ bf16_t hbf[16 * Hz];   // 8KB, rows 8..15 stay zero (M padding)
  __shared__ float  hf32[Bz * Hz];  // 8KB
  __shared__ float  gh[Gz * Bz];    // 24KB: gh[row*8+b]

  const int tid = threadIdx.x;
  const int wave = tid >> 5, lane = tid & 31;
  const int n = lane & 15;
  const bool hi = lane >= 16;
  for (int i = tid; i < 16 * Hz; i += 512) hbf[i] = (bf16_t)0.f;
  for (int i = tid; i < Bz * Hz; i += 512) hf32[i] = 0.f;
  __syncthreads();

  for (int step = 0; step < Sz; ++step) {
    const int s = dir ? (Sz - 1 - step) : step;
    asm volatile("" : "+s"(wp));            // defeat LICM, keep addrspace(1)
    gru_gh_wmma(wp, hbf, gh, wave, n, hi);
    __syncthreads();
    float hn[4];
#pragma unroll
    for (int u = 0; u < 4; ++u) {
      int p = tid * 4 + u; int b = p >> 8; int j = p & 255;
      float sr = gh[j * Bz + b] + bhh[j];
      float sz = gh[(Hz + j) * Bz + b] + bhh[Hz + j];
      float sg = gh[(2 * Hz + j) * Bz + b] + bhh[2 * Hz + j];
      const float* gv = gi + ((size_t)(b * Sz + s)) * Gz;
      float r = sigf(gv[j] + sr);
      float z = sigf(gv[Hz + j] + sz);
      float g = tanhf(gv[2 * Hz + j] + r * sg);
      hn[u] = (1.f - z) * g + z * hf32[b * Hz + j];
    }
    __syncthreads();
#pragma unroll
    for (int u = 0; u < 4; ++u) {
      int p = tid * 4 + u; int b = p >> 8; int j = p & 255;
      hf32[p] = hn[u];
      hbf[b * Hz + j] = (bf16_t)hn[u];
      enc_out[((size_t)(b * Sz + s)) * EOUT + dir * Hz + j] = hn[u];
    }
    __syncthreads();
  }
  for (int i = tid; i < Bz * Hz; i += 512) hfin[i] = hf32[i];
}

// ---------------- enc_state with the reference's (2,B,H)->(B,2H) reshape ----------------
__global__ void enc_state_k(const float* __restrict__ hf, const float* __restrict__ hb,
                            const float* __restrict__ W, const float* __restrict__ bias,
                            float* __restrict__ es) {
  int idx = blockIdx.x * blockDim.x + threadIdx.x;
  if (idx >= Bz * Hz) return;
  int r = idx / Hz, i = idx % Hz;
  const float* src = (r < 4) ? hf : hb;
  int rr = r & 3;
  float s = bias[i];
  const float* wr = W + (size_t)i * (2 * Hz);
  for (int c = 0; c < 2 * Hz; ++c) {
    int bsel = 2 * rr + (c >= Hz);
    s += src[bsel * Hz + (c & (Hz - 1))] * wr[c];
  }
  es[idx] = s;
}

// ---------------- decoder input gates (h-independent, all 64 steps) ----------------
__global__ void dec_gi_k(const float* __restrict__ lab_emb, const float* __restrict__ posf,
                         const float* __restrict__ es, const float* __restrict__ x_emb,
                         const float* __restrict__ Wih_d, const float* __restrict__ bih_d,
                         float* __restrict__ gi_d) {
  const int TOT = Tz * Bz * Gz;
  int idx = blockIdx.x * blockDim.x + threadIdx.x;
  if (idx >= TOT) return;
  int t = idx / (Bz * Gz);
  int rem = idx % (Bz * Gz);
  int b = rem / Gz, g = rem % Gz;
  const float* W = Wih_d + (size_t)g * DEC_IN;
  float s = bih_d[g];
  const float* tgt = lab_emb + ((size_t)(b * Tz + t)) * Ez;
  for (int k = 0; k < Ez; ++k) s += tgt[k] * W[k];
  const float* pf = posf + (size_t)b * Ez;
  for (int k = 0; k < Ez; ++k) s += pf[k] * W[Ez + k];
  const float* eb = es + (size_t)b * Hz;
  for (int k = 0; k < Hz; ++k) s += eb[k] * W[2 * Ez + k];
  const float* xi = x_emb + ((size_t)(b * Sz + t)) * Ez;
  for (int k = 0; k < Ez; ++k) s += xi[k] * W[2 * Ez + Hz + k];
  gi_d[idx] = s;  // layout [t][b][g]
}

// ---------------- decoder recurrence, gates only (attention hoisted) ----------------
__global__ void __launch_bounds__(512) dec_scan_wmma_k(
    const float* __restrict__ gi_d, const bf16_t* __restrict__ Whd,
    const float* __restrict__ bhh_d, bf16_t* __restrict__ featbf,
    float* __restrict__ hseq, float* __restrict__ hT) {
  __shared__ bf16_t hbf[16 * Hz];
  __shared__ float  hf32[Bz * Hz];
  __shared__ float  gh[Gz * Bz];
  const int tid = threadIdx.x;
  const int wave = tid >> 5, lane = tid & 31;
  const int n = lane & 15;
  const bool hi = lane >= 16;
  gbf_ptr wp = (gbf_ptr)Whd;
  for (int i = tid; i < 16 * Hz; i += 512) hbf[i] = (bf16_t)0.f;
  for (int i = tid; i < Bz * Hz; i += 512) hf32[i] = 0.f;
  __syncthreads();

  for (int t = 0; t < Tz; ++t) {
    asm volatile("" : "+s"(wp));
    gru_gh_wmma(wp, hbf, gh, wave, n, hi);
    __syncthreads();
    float hn[4];
#pragma unroll
    for (int u = 0; u < 4; ++u) {
      int p = tid * 4 + u; int b = p >> 8; int j = p & 255;
      float sr = gh[j * Bz + b] + bhh_d[j];
      float sz = gh[(Hz + j) * Bz + b] + bhh_d[Hz + j];
      float sg = gh[(2 * Hz + j) * Bz + b] + bhh_d[2 * Hz + j];
      const float* gv = gi_d + ((size_t)(t * Bz + b)) * Gz;
      float r = sigf(gv[j] + sr);
      float z = sigf(gv[Hz + j] + sz);
      float g = tanhf(gv[2 * Hz + j] + r * sg);
      hn[u] = (1.f - z) * g + z * hf32[b * Hz + j];
    }
    __syncthreads();
#pragma unroll
    for (int u = 0; u < 4; ++u) {
      int p = tid * 4 + u; int b = p >> 8; int j = p & 255;
      hf32[p] = hn[u];
      hbf[b * Hz + j] = (bf16_t)hn[u];
      const int m = t * Bz + b;
      hseq[(size_t)m * Hz + j] = hn[u];
      featbf[(size_t)m * Gz + EOUT + j] = (bf16_t)hn[u];
    }
    __syncthreads();
  }
  for (int i = tid; i < Bz * Hz; i += 512) hT[i] = hf32[i];
}

// ---------------- batched attention (parallel over all 64 steps) ----------------
__global__ void q_k(const float* __restrict__ hseq, const float* __restrict__ W,
                    const float* __restrict__ bias, float* __restrict__ q) {
  int idx = blockIdx.x * blockDim.x + threadIdx.x;
  if (idx >= Mz * EOUT) return;
  int m = idx / EOUT, i = idx % EOUT;
  const float* hr = hseq + (size_t)m * Hz;
  const float* wr = W + (size_t)i * Hz;
  float s = bias[i];
  for (int k = 0; k < Hz; ++k) s += hr[k] * wr[k];
  q[idx] = s;
}
__global__ void attn_k(const float* __restrict__ q, const float* __restrict__ enc_out,
                       float* __restrict__ sc) {
  int idx = blockIdx.x * blockDim.x + threadIdx.x;
  if (idx >= Mz * Sz) return;
  int m = idx / Sz, s_ = idx % Sz;
  int b = m & 7;
  const float* e = enc_out + ((size_t)(b * Sz + s_)) * EOUT;
  const float* qm = q + (size_t)m * EOUT;
  float a = 0.f;
  for (int k = 0; k < EOUT; ++k) a += e[k] * qm[k];
  sc[idx] = a;
}
__global__ void softmax256_k(float* __restrict__ sc) {
  const int row = blockIdx.x;
  float* p = sc + (size_t)row * Sz;
  __shared__ float red[256];
  const int tid = threadIdx.x;
  float v = p[tid];
  red[tid] = v; __syncthreads();
  for (int s = 128; s > 0; s >>= 1) { if (tid < s) red[tid] = fmaxf(red[tid], red[tid + s]); __syncthreads(); }
  const float mx = red[0];
  __syncthreads();
  float e = __expf(v - mx);
  red[tid] = e; __syncthreads();
  for (int s = 128; s > 0; s >>= 1) { if (tid < s) red[tid] += red[tid + s]; __syncthreads(); }
  p[tid] = e / red[0];
}
__global__ void hstar_k(const float* __restrict__ sc, const float* __restrict__ enc_out,
                        bf16_t* __restrict__ featbf) {
  int idx = blockIdx.x * blockDim.x + threadIdx.x;
  if (idx >= Mz * EOUT) return;
  int m = idx / EOUT, d = idx % EOUT;
  int b = m & 7;
  const float* e = enc_out + ((size_t)(b * Sz)) * EOUT + d;
  const float* pm = sc + (size_t)m * Sz;
  float s = 0.f;
  for (int s_ = 0; s_ < Sz; ++s_) s += pm[s_] * e[(size_t)s_ * EOUT];
  featbf[(size_t)m * Gz + d] = (bf16_t)s;
}

// =======================================================================
// Output projection (512x768)@(768x50000): 4Mx4N per wave, ~33 FLOP/B,
// bias+relu fused.  min-waves=1 so the ~200 live VGPRs stay in registers.
// =======================================================================
__global__ void __launch_bounds__(128, 1) logits_wmma_k(
    const bf16_t* __restrict__ featbf, const bf16_t* __restrict__ Wbf,
    const float* __restrict__ bias, float* __restrict__ out) {
  const int NT = Vz / 16;  // 3125 column tiles
  const int wave = threadIdx.x >> 5;
  const int lane = threadIdx.x & 31;
  const int ng0 = (blockIdx.x * 4 + wave) * 4;   // first of 4 n-tiles
  if (ng0 >= NT) return;                         // wave-uniform
  const int m0 = blockIdx.y * 64;                // 4 m-tiles
  const int n = lane & 15;
  const bool hi = lane >= 16;
  v8f c[4][4] = {};
  for (int kb = 0; kb < Gz / 32; ++kb) {
    const int k0 = kb * 32;
    v16bf a[4], b[4];
#pragma unroll
    for (int mt = 0; mt < 4; ++mt)
      a[mt] = load_frag_a(featbf + (size_t)(m0 + mt * 16 + n) * Gz + k0 + (hi ? 16 : 0));
#pragma unroll
    for (int nt = 0; nt < 4; ++nt)
      if (ng0 + nt < NT) {
        const bf16_t* bp = Wbf + (size_t)((ng0 + nt) * 16 + n) * Gz + k0 + (hi ? 16 : 0);
        __builtin_prefetch(bp + 32, 0, 1);
        b[nt] = load_frag_b(bp);
      }
#pragma unroll
    for (int mt = 0; mt < 4; ++mt)
#pragma unroll
      for (int nt = 0; nt < 4; ++nt)
        if (ng0 + nt < NT)
          c[mt][nt] = __builtin_amdgcn_wmma_f32_16x16x32_bf16(false, a[mt], false, b[nt],
                                                              (short)0, c[mt][nt], false, false);
  }
#pragma unroll
  for (int nt = 0; nt < 4; ++nt) {
    if (ng0 + nt >= NT) continue;
    const int vcol = (ng0 + nt) * 16 + n;
    const float bb = bias[vcol];
#pragma unroll
    for (int mt = 0; mt < 4; ++mt)
#pragma unroll
      for (int r = 0; r < 8; ++r) {
        int m = m0 + mt * 16 + r + (hi ? 8 : 0);
        float val = c[mt][nt][r] + bb;
        val = val < 0.f ? 0.f : val;
        out[(size_t)m * Vz + vcol] = val;
      }
  }
}

// ---------------- in-place log_softmax over V per row ----------------
__global__ void logsoftmax_k(float* __restrict__ data) {
  const int row = blockIdx.x;
  float* p = data + (size_t)row * Vz;
  __shared__ float red[256];
  const int tid = threadIdx.x;
  float m = -3.4e38f;
  for (int i = tid; i < Vz; i += 256) m = fmaxf(m, p[i]);
  red[tid] = m; __syncthreads();
  for (int s = 128; s > 0; s >>= 1) { if (tid < s) red[tid] = fmaxf(red[tid], red[tid + s]); __syncthreads(); }
  const float mx = red[0];
  __syncthreads();
  float sum = 0.f;
  for (int i = tid; i < Vz; i += 256) sum += __expf(p[i] - mx);
  red[tid] = sum; __syncthreads();
  for (int s = 128; s > 0; s >>= 1) { if (tid < s) red[tid] += red[tid + s]; __syncthreads(); }
  const float lz = __logf(red[0]) + mx;
  for (int i = tid; i < Vz; i += 256) p[i] = p[i] - lz;
}

// =======================================================================
extern "C" void kernel_launch(void* const* d_in, const int* in_sizes, int n_in,
                              void* d_out, int out_size, void* d_ws, size_t ws_size,
                              hipStream_t stream) {
  (void)in_sizes; (void)n_in; (void)out_size; (void)ws_size;
  const int*   input     = (const int*)  d_in[0];
  const int*   pos       = (const int*)  d_in[1];
  const int*   label     = (const int*)  d_in[2];
  const float* char_emb  = (const float*)d_in[3];
  const float* pos_emb   = (const float*)d_in[4];
  const float* Wih_f     = (const float*)d_in[5];
  const float* Whh_f     = (const float*)d_in[6];
  const float* bih_f     = (const float*)d_in[7];
  const float* bhh_f     = (const float*)d_in[8];
  const float* Wih_b     = (const float*)d_in[9];
  const float* Whh_b     = (const float*)d_in[10];
  const float* bih_b     = (const float*)d_in[11];
  const float* bhh_b     = (const float*)d_in[12];
  const float* fc_h_w    = (const float*)d_in[13];
  const float* fc_h_b    = (const float*)d_in[14];
  const float* fc_pos1_w = (const float*)d_in[15];
  const float* fc_pos1_b = (const float*)d_in[16];
  const float* fc_pos2_w = (const float*)d_in[17];
  const float* fc_pos2_b = (const float*)d_in[18];
  const float* Wih_d     = (const float*)d_in[19];
  const float* Whh_d     = (const float*)d_in[20];
  const float* bih_d     = (const float*)d_in[21];
  const float* bhh_d     = (const float*)d_in[22];
  const float* line_in_w = (const float*)d_in[23];
  const float* line_in_b = (const float*)d_in[24];
  const float* line_out_w= (const float*)d_in[25];
  const float* line_out_b= (const float*)d_in[26];

  // ---- workspace carve-up (256B aligned) ----
  char* base = (char*)d_ws;
  size_t off = 0;
  auto alloc = [&](size_t bytes) -> char* {
    char* p = base + off;
    off = (off + bytes + 255) & ~(size_t)255;
    return p;
  };
  float*  x_emb   = (float*) alloc((size_t)Bz * Sz * Ez * 4);
  float*  lab_emb = (float*) alloc((size_t)Bz * Tz * Ez * 4);
  float*  pos_e   = (float*) alloc((size_t)Bz * POS_SZ * 4);
  float*  f1      = (float*) alloc((size_t)Bz * POS_H * 4);
  float*  posf    = (float*) alloc((size_t)Bz * Ez * 4);
  float*  gi_f    = (float*) alloc((size_t)Bz * Sz * Gz * 4);
  float*  gi_b    = (float*) alloc((size_t)Bz * Sz * Gz * 4);
  float*  enc_out = (float*) alloc((size_t)Bz * Sz * EOUT * 4);
  float*  hf      = (float*) alloc((size_t)Bz * Hz * 4);
  float*  hb      = (float*) alloc((size_t)Bz * Hz * 4);
  float*  es      = (float*) alloc((size_t)Bz * Hz * 4);
  float*  gi_d    = (float*) alloc((size_t)Tz * Bz * Gz * 4);
  float*  hseq    = (float*) alloc((size_t)Mz * Hz * 4);
  float*  qbuf    = (float*) alloc((size_t)Mz * EOUT * 4);
  float*  scbuf   = (float*) alloc((size_t)Mz * Sz * 4);
  bf16_t* featbf  = (bf16_t*)alloc((size_t)Mz * Gz * 2);
  bf16_t* Whf_bf  = (bf16_t*)alloc((size_t)Gz * Hz * 2);
  bf16_t* Whb_bf  = (bf16_t*)alloc((size_t)Gz * Hz * 2);
  bf16_t* Whd_bf  = (bf16_t*)alloc((size_t)Gz * Hz * 2);
  bf16_t* Wbf     = (bf16_t*)alloc((size_t)Vz * Gz * 2);   // 77 MB

  float* hT_out   = (float*)d_out;                 // (B,H)
  float* outs     = (float*)d_out + Bz * Hz;       // (T,B,V)

  // 1) gathers
  gather_k<<<(Bz * Sz * Ez + 255) / 256, 256, 0, stream>>>(input, char_emb, x_emb, Bz * Sz, Ez);
  gather_k<<<(Bz * Tz * Ez + 255) / 256, 256, 0, stream>>>(label, char_emb, lab_emb, Bz * Tz, Ez);
  gather_k<<<(Bz * MPz * Ez + 255) / 256, 256, 0, stream>>>(pos, pos_emb, pos_e, Bz * MPz, Ez);

  // 2) bf16 weight conversions
  tobf16_k<<<4096, 256, 0, stream>>>(line_out_w, Wbf, (long long)Vz * Gz);
  tobf16_k<<<256, 256, 0, stream>>>(Whh_f, Whf_bf, (long long)Gz * Hz);
  tobf16_k<<<256, 256, 0, stream>>>(Whh_b, Whb_bf, (long long)Gz * Hz);
  tobf16_k<<<256, 256, 0, stream>>>(Whh_d, Whd_bf, (long long)Gz * Hz);

  // 3) position MLP
  fc_tanh_k<<<(Bz * POS_H + 255) / 256, 256, 0, stream>>>(pos_e, fc_pos1_w, fc_pos1_b, f1, POS_SZ, POS_H, Bz * POS_H);
  fc_tanh_k<<<(Bz * Ez + 255) / 256, 256, 0, stream>>>(f1, fc_pos2_w, fc_pos2_b, posf, POS_H, Ez, Bz * Ez);

  // 4) encoder input gates
  enc_gi_k<<<(2 * Bz * Sz * Gz + 255) / 256, 256, 0, stream>>>(x_emb, Wih_f, bih_f, Wih_b, bih_b, gi_f, gi_b);

  // 5) encoder recurrences (WMMA in the scan), fwd/bwd on 2 WGPs
  enc_scan_wmma_k<<<2, 512, 0, stream>>>(gi_f, gi_b, Whf_bf, Whb_bf, bhh_f, bhh_b, enc_out, hf, hb);

  // 6) enc_state
  enc_state_k<<<(Bz * Hz + 255) / 256, 256, 0, stream>>>(hf, hb, fc_h_w, fc_h_b, es);

  // 7) decoder input gates (hoisted)
  dec_gi_k<<<(Tz * Bz * Gz + 255) / 256, 256, 0, stream>>>(lab_emb, posf, es, x_emb, Wih_d, bih_d, gi_d);

  // 8) decoder recurrence, gates only (WMMA in the scan)
  dec_scan_wmma_k<<<1, 512, 0, stream>>>(gi_d, Whd_bf, bhh_d, featbf, hseq, hT_out);

  // 9) attention, batched over all 64 steps (fully parallel)
  q_k<<<(Mz * EOUT + 255) / 256, 256, 0, stream>>>(hseq, line_in_w, line_in_b, qbuf);
  attn_k<<<(Mz * Sz + 255) / 256, 256, 0, stream>>>(qbuf, enc_out, scbuf);
  softmax256_k<<<Mz, 256, 0, stream>>>(scbuf);
  hstar_k<<<(Mz * EOUT + 255) / 256, 256, 0, stream>>>(scbuf, enc_out, featbf);

  // 10) output projection: 4Mx4N register-blocked WMMA, bias+relu fused
  dim3 ggrid((Vz / 16 + 15) / 16, (Mz / 64));   // (196, 8), 128 thr
  logits_wmma_k<<<ggrid, 128, 0, stream>>>(featbf, Wbf, line_out_b, outs);

  // 11) in-place log_softmax
  logsoftmax_k<<<Mz, 256, 0, stream>>>(outs);
}
